// IMGCN_79748952752769
// MI455X (gfx1250) — compile-verified, matrix operand order checked by reading, not run on hardware
//
#include <hip/hip_runtime.h>
#include <hip/hip_bf16.h>
#include <math.h>

// ---------------------------------------------------------------------------
// IMGCN forward for MI455X (gfx1250, wave32, WMMA).
// Heavy contractions (fc_in, adj@z, Fourier-KAN features) use
// v_wmma_f32_16x16x32_bf16 with on-the-fly f32->bf16 conversion (f32 accum).
// adj streaming (2x268MB) is the bandwidth bottleneck -> contiguous loads +
// prefetch hints; all K-tail handling is branchless (clamp + select).
// ---------------------------------------------------------------------------

#define NN     8192
#define INCH   2000
#define HIDD   32
#define GRIDF  200
#define OUTCH  16
#define NM     (NN * HIDD)          // 262144 floats per [N,32] buffer

// workspace buffer offsets (floats)
#define WS_H    (0 * NM)
#define WS_Q    (1 * NM)
#define WS_K    (2 * NM)
#define WS_V    (3 * NM)
#define WS_ATTN (4 * NM)
#define WS_Z    (5 * NM)            // also reused as pre-LN h from fc_in
#define WS_AZ   (6 * NM)
#define WS_Y    (7 * NM)
#define WS_ZT   (8 * NM)            // z transposed [32, NN]
#define WS_SC   (9 * NM)            // scalar/reduction block (2048 floats)

// offsets inside scalar block
#define OFF_SSQ  0
#define OFF_SSK  1
#define OFF_KSUM 2
#define OFF_VSUM 34
#define OFF_KVS  66
#define OFF_YSUM 1090
#define OFF_MEAN 1122
#define SC_CLEAR 2048

typedef __attribute__((ext_vector_type(16))) __bf16 bf16x16;
typedef __attribute__((ext_vector_type(8)))  float  f32x8;

// K-index contributed by element e (0..15) of a lane's bf16 A/B fragment for a
// K=32 16-bit tile (ISA 7.12.2): VGPR0..3 hold K 0..7 (half0) / 8..15 (half1),
// VGPR4..7 hold K 16..23 (half0) / 24..31 (half1).
__device__ __forceinline__ int kmap(int e, int half) {
  return ((e < 8) ? e : (e + 8)) + (half << 3);
}

__device__ __forceinline__ f32x8 wmma_bf16(bf16x16 a, bf16x16 b, f32x8 c) {
  return __builtin_amdgcn_wmma_f32_16x16x32_bf16(false, a, false, b,
                                                 (short)0, c, false, false);
}

// ---------------------------------------------------------------------------
// mask = sigmoid(mask_w) broadcast to [N, INCH]
__global__ void mask_kernel(const float* __restrict__ mw, float* __restrict__ out) {
  long idx = (long)blockIdx.x * blockDim.x + threadIdx.x;
  if (idx >= (long)NN * INCH) return;
  int j = (int)(idx % INCH);
  out[idx] = 1.0f / (1.0f + __expf(-mw[j]));
}

// zero a float range
__global__ void zero_kernel(float* __restrict__ p, int n) {
  int i = blockIdx.x * blockDim.x + threadIdx.x;
  if (i < n) p[i] = 0.0f;
}

// ---------------------------------------------------------------------------
// hraw[n,j] = sum_c x[c,n] * Win[j,c]  + bias  (M=8192, N=32, K=2000)
// one wave per 16-row tile, both 16-wide column tiles; 512 waves.
__global__ void fcin_wmma(const float* __restrict__ x, const float* __restrict__ win,
                          const float* __restrict__ bias, float* __restrict__ hraw) {
  int gw   = blockIdx.x * (blockDim.x >> 5) + (threadIdx.x >> 5);
  int n0   = gw * 16;
  int lane = threadIdx.x & 31;
  int half = lane >> 4, m = lane & 15;
  const float* wa = win + (long)m * INCH;          // column j = m
  const float* wb = win + (long)(16 + m) * INCH;   // column j = 16+m
  f32x8 acc0 = {0.f,0.f,0.f,0.f,0.f,0.f,0.f,0.f};
  f32x8 acc1 = acc0;
  // 62 full K-blocks (unguarded)
  for (int kb = 0; kb < 1984; kb += 32) {
    bf16x16 a, b0, b1;
#pragma unroll
    for (int e = 0; e < 16; ++e) {
      int k = kb + kmap(e, half);
      a[e]  = (__bf16)x[(long)k * NN + n0 + m];
      b0[e] = (__bf16)wa[k];
      b1[e] = (__bf16)wb[k];
    }
    acc0 = wmma_bf16(a, b0, acc0);
    acc1 = wmma_bf16(a, b1, acc1);
  }
  // tail block: k in [1984,2016), valid < 2000 -> clamped loads + value select
  {
    bf16x16 a, b0, b1;
#pragma unroll
    for (int e = 0; e < 16; ++e) {
      int k  = 1984 + kmap(e, half);
      int kc = (k < INCH) ? k : (INCH - 1);
      float xv = x[(long)kc * NN + n0 + m];
      float w0 = wa[kc];
      float w1 = wb[kc];
      bool ok = (k < INCH);
      a[e]  = (__bf16)(ok ? xv : 0.0f);
      b0[e] = (__bf16)(ok ? w0 : 0.0f);
      b1[e] = (__bf16)(ok ? w1 : 0.0f);
    }
    acc0 = wmma_bf16(a, b0, acc0);
    acc1 = wmma_bf16(a, b1, acc1);
  }
  float ba = bias[m], bb = bias[16 + m];
#pragma unroll
  for (int r = 0; r < 8; ++r) {
    int row = n0 + r + half * 8;                   // C/D layout: VGPR r -> M=r(+8)
    hraw[(long)row * HIDD + m]      = acc0[r] + ba;
    hraw[(long)row * HIDD + 16 + m] = acc1[r] + bb;
  }
}

// ---------------------------------------------------------------------------
// zt[j, n] = z[n, j]  (32 x NN), LDS tile transpose (padded stride: no bank
// conflicts). 256 threads handle a 32x32 tile.
__global__ void transpose32_kernel(const float* __restrict__ z, float* __restrict__ zt) {
  __shared__ float tile[32][33];
  int n0  = blockIdx.x * 32;
  int tid = threadIdx.x;
  int r   = tid >> 5;          // 0..7
  int c   = tid & 31;
#pragma unroll
  for (int rr = 0; rr < 32; rr += 8)
    tile[r + rr][c] = z[(long)(n0 + r + rr) * HIDD + c];
  __syncthreads();
#pragma unroll
  for (int jj = 0; jj < 32; jj += 8)
    zt[(long)(r + jj) * NN + n0 + c] = tile[c][r + jj];
}

// ---------------------------------------------------------------------------
// az = adj @ z  (M=8192, N=32, K=8192).  A rows contiguous; B from transposed
// z so each lane reads contiguous 32B runs (b128-mergeable).
// one wave per 16-row tile handles both column tiles; 512 waves.
__global__ void adjz_wmma(const float* __restrict__ adj, const float* __restrict__ zt,
                          float* __restrict__ az) {
  int gw   = blockIdx.x * (blockDim.x >> 5) + (threadIdx.x >> 5);
  int n0   = gw * 16;
  int lane = threadIdx.x & 31;
  int half = lane >> 4, m = lane & 15;
  const float* arow = adj + (long)(n0 + m) * NN;
  const float* zta  = zt + (long)m * NN;           // column j = m
  const float* ztb  = zt + (long)(16 + m) * NN;    // column j = 16+m
  f32x8 acc0 = {0.f,0.f,0.f,0.f,0.f,0.f,0.f,0.f};
  f32x8 acc1 = acc0;
  for (int kb = 0; kb < NN; kb += 32) {
    if (kb + 1024 < NN) __builtin_prefetch(arow + kb + 1024, 0, 0);  // global_prefetch_b8
    bf16x16 a, b0, b1;
#pragma unroll
    for (int e = 0; e < 16; ++e) {
      int k = kb + kmap(e, half);
      a[e]  = (__bf16)arow[k];
      b0[e] = (__bf16)zta[k];
      b1[e] = (__bf16)ztb[k];
    }
    acc0 = wmma_bf16(a, b0, acc0);
    acc1 = wmma_bf16(a, b1, acc1);
  }
#pragma unroll
  for (int r = 0; r < 8; ++r) {
    int row = n0 + r + half * 8;
    az[(long)row * HIDD + m]      = acc0[r];
    az[(long)row * HIDD + 16 + m] = acc1[r];
  }
}

// ---------------------------------------------------------------------------
// Fourier-KAN: y[n,j] = sum_{i,k} cos(az[n,i]*(k+1))*C0[j,i,k]
//                                + sin(az[n,i]*(k+1))*C1[j,i,k]  + kbias[j]
// Implicit GEMM, K = 32*200 per trig branch; features built in-register.
// one wave per 16-row tile, both column tiles; 512 waves.
__global__ void kan_wmma(const float* __restrict__ az, const float* __restrict__ C0,
                         const float* __restrict__ C1, const float* __restrict__ kbias,
                         float* __restrict__ y) {
  int gw   = blockIdx.x * (blockDim.x >> 5) + (threadIdx.x >> 5);
  int n0   = gw * 16;
  int lane = threadIdx.x & 31;
  int half = lane >> 4, m = lane & 15;
  f32x8 acc0 = {0.f,0.f,0.f,0.f,0.f,0.f,0.f,0.f};
  f32x8 acc1 = acc0;
  for (int i = 0; i < HIDD; ++i) {
    float av = az[(long)(n0 + m) * HIDD + i];      // this lane's row value
    const float* c0a = C0 + ((long)m * HIDD + i) * GRIDF;         // j = m
    const float* c0b = C0 + ((long)(16 + m) * HIDD + i) * GRIDF;  // j = 16+m
    const float* c1a = C1 + ((long)m * HIDD + i) * GRIDF;
    const float* c1b = C1 + ((long)(16 + m) * HIDD + i) * GRIDF;
    // 6 full K-blocks (unguarded): grid k in [0,192)
    for (int kb = 0; kb < 192; kb += 32) {
      bf16x16 ac, as, bc0, bc1, bs0, bs1;
#pragma unroll
      for (int e = 0; e < 16; ++e) {
        int kl = kb + kmap(e, half);
        float sv, cv;
        __sincosf(av * (float)(kl + 1), &sv, &cv);
        ac[e]  = (__bf16)cv;
        as[e]  = (__bf16)sv;
        bc0[e] = (__bf16)c0a[kl];
        bc1[e] = (__bf16)c0b[kl];
        bs0[e] = (__bf16)c1a[kl];
        bs1[e] = (__bf16)c1b[kl];
      }
      acc0 = wmma_bf16(ac, bc0, acc0);
      acc0 = wmma_bf16(as, bs0, acc0);
      acc1 = wmma_bf16(ac, bc1, acc1);
      acc1 = wmma_bf16(as, bs1, acc1);
    }
    // tail block: grid k in [192,224), valid < 200 -> clamp + select
    {
      bf16x16 ac, as, bc0, bc1, bs0, bs1;
#pragma unroll
      for (int e = 0; e < 16; ++e) {
        int kl = 192 + kmap(e, half);
        int kc = (kl < GRIDF) ? kl : (GRIDF - 1);
        bool ok = (kl < GRIDF);
        float sv, cv;
        __sincosf(av * (float)(kc + 1), &sv, &cv);
        float v0 = c0a[kc], v1 = c0b[kc], v2 = c1a[kc], v3 = c1b[kc];
        ac[e]  = (__bf16)(ok ? cv : 0.0f);
        as[e]  = (__bf16)(ok ? sv : 0.0f);
        bc0[e] = (__bf16)(ok ? v0 : 0.0f);
        bc1[e] = (__bf16)(ok ? v1 : 0.0f);
        bs0[e] = (__bf16)(ok ? v2 : 0.0f);
        bs1[e] = (__bf16)(ok ? v3 : 0.0f);
      }
      acc0 = wmma_bf16(ac, bc0, acc0);
      acc0 = wmma_bf16(as, bs0, acc0);
      acc1 = wmma_bf16(ac, bc1, acc1);
      acc1 = wmma_bf16(as, bs1, acc1);
    }
  }
  float ba = kbias[m], bb = kbias[16 + m];
#pragma unroll
  for (int r = 0; r < 8; ++r) {
    int row = n0 + r + half * 8;
    y[(long)row * HIDD + m]      = acc0[r] + ba;
    y[(long)row * HIDD + 16 + m] = acc1[r] + bb;
  }
}

// ---------------------------------------------------------------------------
// out[n,j] = sum_p in[n,p]*w[j,p] + b[j]   (w: [32,32])  -- q/k/v/kan_in
__global__ void lin32_kernel(const float* __restrict__ in, const float* __restrict__ w,
                             const float* __restrict__ b, float* __restrict__ out) {
  __shared__ float sh[8][HIDD];
  int tid  = threadIdx.x;
  int node = blockIdx.x * 8 + (tid >> 5);
  int j    = tid & 31;
  sh[tid >> 5][j] = in[(long)node * HIDD + j];
  __syncthreads();
  const float* hr = sh[tid >> 5];
  float acc = b[j];
#pragma unroll
  for (int p = 0; p < HIDD; ++p) acc = fmaf(hr[p], w[j * HIDD + p], acc);
  out[(long)node * HIDD + j] = acc;
}

// LayerNorm(+optional relu) of src row -> dst row
__global__ void ln_kernel(const float* __restrict__ src, const float* __restrict__ g,
                          const float* __restrict__ b, float* __restrict__ dst, int do_relu) {
  int n = blockIdx.x * blockDim.x + threadIdx.x;
  if (n >= NN) return;
  float v[HIDD];
  float mean = 0.0f;
#pragma unroll
  for (int j = 0; j < HIDD; ++j) { v[j] = src[(long)n * HIDD + j]; mean += v[j]; }
  mean *= (1.0f / HIDD);
  float var = 0.0f;
#pragma unroll
  for (int j = 0; j < HIDD; ++j) { float d = v[j] - mean; var += d * d; }
  var *= (1.0f / HIDD);
  float rstd = rsqrtf(var + 1e-5f);
#pragma unroll
  for (int j = 0; j < HIDD; ++j) {
    float o = (v[j] - mean) * rstd * g[j] + b[j];
    dst[(long)n * HIDD + j] = do_relu ? fmaxf(o, 0.0f) : o;
  }
}

// ssq(q), ssk(k), ksum[32], vsum[32] via LDS partials (ds_add_f32) + atomics
__global__ void reduce1_kernel(const float* __restrict__ q, const float* __restrict__ k,
                               const float* __restrict__ v, float* __restrict__ sc) {
  __shared__ float lks[HIDD], lvs[HIDD], lsq, lsk;
  int tid = threadIdx.x;
  if (tid < HIDD) { lks[tid] = 0.0f; lvs[tid] = 0.0f; }
  if (tid == 0)   { lsq = 0.0f; lsk = 0.0f; }
  __syncthreads();
  int n = blockIdx.x * blockDim.x + tid;      // exactly NN threads total
  float pq = 0.0f, pk = 0.0f;
#pragma unroll 4
  for (int j = 0; j < HIDD; ++j) {
    float qv = q[(long)n * HIDD + j];
    float kv = k[(long)n * HIDD + j];
    float vv = v[(long)n * HIDD + j];
    pq = fmaf(qv, qv, pq);
    pk = fmaf(kv, kv, pk);
    atomicAdd(&lks[j], kv);
    atomicAdd(&lvs[j], vv);
  }
  atomicAdd(&lsq, pq);
  atomicAdd(&lsk, pk);
  __syncthreads();
  if (tid < HIDD) {
    atomicAdd(&sc[OFF_KSUM + tid], lks[tid]);
    atomicAdd(&sc[OFF_VSUM + tid], lvs[tid]);
  }
  if (tid == 0) { atomicAdd(&sc[OFF_SSQ], lsq); atomicAdd(&sc[OFF_SSK], lsk); }
}

// kvs[m,d] = sum_n k[n,m]*v[n,d]; 1024 threads/block, one accumulator each
__global__ void kvs_kernel(const float* __restrict__ k, const float* __restrict__ v,
                           float* __restrict__ sc) {
  int t = threadIdx.x;                         // 1024
  int mrow = t >> 5, d = t & 31;
  int n0 = blockIdx.x * 1024;
  float acc = 0.0f;
  for (int n = n0; n < n0 + 1024; ++n)
    acc = fmaf(k[(long)n * HIDD + mrow], v[(long)n * HIDD + d], acc);
  atomicAdd(&sc[OFF_KVS + t], acc);
}

// attn[n,d] = (q[n]·kvs[:,d]·rr + vsum[d]) / (q[n]·ksum·rr + N)
__global__ void attn_kernel(const float* __restrict__ q, const float* __restrict__ sc,
                            float* __restrict__ attn) {
  int idx = blockIdx.x * blockDim.x + threadIdx.x;
  int n = idx >> 5, d = idx & 31;
  float rr = rsqrtf(sc[OFF_SSQ]) * rsqrtf(sc[OFF_SSK]);
  float num = 0.0f, den = 0.0f;
#pragma unroll
  for (int p = 0; p < HIDD; ++p) {
    float qv = q[(long)n * HIDD + p];
    num = fmaf(qv, sc[OFF_KVS + p * HIDD + d], num);
    den = fmaf(qv, sc[OFF_KSUM + p], den);
  }
  attn[(long)n * HIDD + d] = (num * rr + sc[OFF_VSUM + d]) / (den * rr + (float)NN);
}

// column sums of y -> sc[OFF_YSUM..]
__global__ void colsum_kernel(const float* __restrict__ y, float* __restrict__ sc) {
  __shared__ float ls[HIDD];
  int tid = threadIdx.x;
  if (tid < HIDD) ls[tid] = 0.0f;
  __syncthreads();
  int n = blockIdx.x * blockDim.x + tid;
#pragma unroll 4
  for (int j = 0; j < HIDD; ++j) atomicAdd(&ls[j], y[(long)n * HIDD + j]);
  __syncthreads();
  if (tid < HIDD) atomicAdd(&sc[OFF_YSUM + tid], ls[tid]);
}

// meanvec[j] = (ysum · w_out[j,:]) / N    (kan_out has no bias)
__global__ void meanvec_kernel(const float* __restrict__ wout, float* __restrict__ sc) {
  int j = threadIdx.x;                         // 32 threads
  float acc = 0.0f;
#pragma unroll
  for (int p = 0; p < HIDD; ++p) acc = fmaf(sc[OFF_YSUM + p], wout[j * HIDD + p], acc);
  sc[OFF_MEAN + j] = acc * (1.0f / (float)NN);
}

// h = LN(0.5*(attn + meanvec) + 0.5*h)  (in place on h)
__global__ void update_kernel(const float* __restrict__ attn, const float* __restrict__ sc,
                              const float* __restrict__ g, const float* __restrict__ b,
                              float* __restrict__ h) {
  int n = blockIdx.x * blockDim.x + threadIdx.x;
  if (n >= NN) return;
  float v[HIDD];
  float mean = 0.0f;
#pragma unroll
  for (int j = 0; j < HIDD; ++j) {
    v[j] = 0.5f * (attn[(long)n * HIDD + j] + sc[OFF_MEAN + j])
         + 0.5f * h[(long)n * HIDD + j];
    mean += v[j];
  }
  mean *= (1.0f / HIDD);
  float var = 0.0f;
#pragma unroll
  for (int j = 0; j < HIDD; ++j) { float d = v[j] - mean; var += d * d; }
  var *= (1.0f / HIDD);
  float rstd = rsqrtf(var + 1e-5f);
#pragma unroll
  for (int j = 0; j < HIDD; ++j)
    h[(long)n * HIDD + j] = (v[j] - mean) * rstd * g[j] + b[j];
}

// x_out[n,o] = h[n]·w[o,:] + b[o]
__global__ void fcout_kernel(const float* __restrict__ h, const float* __restrict__ w,
                             const float* __restrict__ b, float* __restrict__ out) {
  int idx = blockIdx.x * blockDim.x + threadIdx.x;
  int n = idx >> 4, o = idx & 15;
  float acc = b[o];
#pragma unroll
  for (int p = 0; p < HIDD; ++p) acc = fmaf(h[(long)n * HIDD + p], w[o * HIDD + p], acc);
  out[(long)n * OUTCH + o] = acc;
}

// ---------------------------------------------------------------------------
extern "C" void kernel_launch(void* const* d_in, const int* in_sizes, int n_in,
                              void* d_out, int out_size, void* d_ws, size_t ws_size,
                              hipStream_t stream) {
  const float* x        = (const float*)d_in[0];
  const float* adj      = (const float*)d_in[1];
  const float* mask_w   = (const float*)d_in[2];
  const float* fc_in_w  = (const float*)d_in[3];
  const float* fc_in_b  = (const float*)d_in[4];
  const float* ln_g     = (const float*)d_in[5];
  const float* ln_b     = (const float*)d_in[6];
  const float* Wq_w     = (const float*)d_in[7];
  const float* Wq_b     = (const float*)d_in[8];
  const float* Wk_w     = (const float*)d_in[9];
  const float* Wk_b     = (const float*)d_in[10];
  const float* Wv_w     = (const float*)d_in[11];
  const float* Wv_b     = (const float*)d_in[12];
  const float* kan_in_w = (const float*)d_in[13];
  const float* kan_in_b = (const float*)d_in[14];
  const float* fourier  = (const float*)d_in[15];
  const float* kan_bias = (const float*)d_in[16];
  const float* kan_out_w= (const float*)d_in[17];
  const float* fc_out_w = (const float*)d_in[18];
  const float* fc_out_b = (const float*)d_in[19];

  float* ws  = (float*)d_ws;
  float* out = (float*)d_out;
  float* out_mask = out + (long)NN * OUTCH;

  float* H  = ws + WS_H;
  float* Q  = ws + WS_Q;
  float* K  = ws + WS_K;
  float* V  = ws + WS_V;
  float* AT = ws + WS_ATTN;
  float* Z  = ws + WS_Z;
  float* AZ = ws + WS_AZ;
  float* Y  = ws + WS_Y;
  float* ZT = ws + WS_ZT;
  float* SC = ws + WS_SC;

  // mask output (independent)
  {
    long total = (long)NN * INCH;
    mask_kernel<<<(int)((total + 255) / 256), 256, 0, stream>>>(mask_w, out_mask);
  }

  // h = relu(LN(x^T @ fc_in_w^T + b))
  fcin_wmma<<<64, 256, 0, stream>>>(x, fc_in_w, fc_in_b, Z);      // 512 waves
  ln_kernel<<<32, 256, 0, stream>>>(Z, ln_g, ln_b, H, 1);

  for (int i = 0; i < 2; ++i) {
    zero_kernel<<<(SC_CLEAR + 255) / 256, 256, 0, stream>>>(SC, SC_CLEAR);

    lin32_kernel<<<1024, 256, 0, stream>>>(H, Wq_w + i * 1024, Wq_b + i * 32, Q);
    lin32_kernel<<<1024, 256, 0, stream>>>(H, Wk_w + i * 1024, Wk_b + i * 32, K);
    lin32_kernel<<<1024, 256, 0, stream>>>(H, Wv_w + i * 1024, Wv_b + i * 32, V);

    reduce1_kernel<<<32, 256, 0, stream>>>(Q, K, V, SC);
    kvs_kernel<<<8, 1024, 0, stream>>>(K, V, SC);
    attn_kernel<<<1024, 256, 0, stream>>>(Q, SC, AT);

    lin32_kernel<<<1024, 256, 0, stream>>>(H, kan_in_w + i * 1024, kan_in_b + i * 32, Z);
    transpose32_kernel<<<NN / 32, 256, 0, stream>>>(Z, ZT);
    adjz_wmma<<<64, 256, 0, stream>>>(adj, ZT, AZ);               // 512 waves

    const float* C0 = fourier + (long)i * 2 * HIDD * HIDD * GRIDF;
    const float* C1 = C0 + (long)HIDD * HIDD * GRIDF;
    kan_wmma<<<64, 256, 0, stream>>>(AZ, C0, C1, kan_bias + i * 32, Y);

    colsum_kernel<<<32, 256, 0, stream>>>(Y, SC);
    meanvec_kernel<<<1, 32, 0, stream>>>(kan_out_w + i * 1024, SC);

    update_kernel<<<32, 256, 0, stream>>>(AT, SC, ln_g + (i + 1) * 32,
                                          ln_b + (i + 1) * 32, H);
  }

  fcout_kernel<<<512, 256, 0, stream>>>(H, fc_out_w, fc_out_b, out);
}
